// MambaTimeSeries_12807592477040
// MI455X (gfx1250) — matrix operand order repaired and checked by reading, not compiled, not run on hardware
//
#include <hip/hip_runtime.h>
#include <hip/hip_bf16.h>

// ---------------------------------------------------------------------------
// Problem constants (from reference): B=32, L=4096, D=512, K=4, N=4 layers
// ---------------------------------------------------------------------------
#define BB   32
#define LL   4096
#define DD   512
#define KK   4
#define NL   4
#define MROWS (BB * LL)          // 131072 rows of [*, D]
#define EPSF 1e-5f

typedef __attribute__((ext_vector_type(16))) __bf16 v16bf;
typedef __attribute__((ext_vector_type(8)))  __bf16 v8bf;
typedef __attribute__((ext_vector_type(4)))  __bf16 v4bf;
typedef __attribute__((ext_vector_type(8)))  float  v8f;

// ---------------------------------------------------------------------------
// bf16 helpers (round-to-nearest-even)
// ---------------------------------------------------------------------------
__device__ __forceinline__ __bf16 f2bf(float f) {
    union { float f; unsigned u; } v; v.f = f;
    unsigned r = v.u + 0x7FFFu + ((v.u >> 16) & 1u);
    unsigned short h = (unsigned short)(r >> 16);
    return __builtin_bit_cast(__bf16, h);
}
__device__ __forceinline__ float bf2f(__bf16 b) {
    unsigned short h = __builtin_bit_cast(unsigned short, b);
    union { unsigned u; float f; } v; v.u = ((unsigned)h) << 16;
    return v.f;
}

// ---------------------------------------------------------------------------
// Async global -> LDS copy of 32 bytes (16 bf16) per lane, no VGPR staging.
// gfx1250: tracked by ASYNCcnt; IOFFSET applies to BOTH lds and global addr.
// Generic pointers to LDS carry the LDS byte address in their low 32 bits
// (ISA flat aperture).
// ---------------------------------------------------------------------------
__device__ __forceinline__ void async_ld32B(__bf16* ldsp, const __bf16* gp) {
    unsigned l = (unsigned)(unsigned long long)ldsp;
    asm volatile("global_load_async_to_lds_b128 %0, %1, off\n\t"
                 "global_load_async_to_lds_b128 %0, %1, off offset:16"
                 :: "v"(l), "v"(gp) : "memory");
}
__device__ __forceinline__ void wait_async0() {
    asm volatile("s_wait_asynccnt 0" ::: "memory");
}

// ---------------------------------------------------------------------------
// f32 -> bf16 weight conversion
// ---------------------------------------------------------------------------
__global__ void __launch_bounds__(256)
cvt_bf16_kernel(const float* __restrict__ in, __bf16* __restrict__ out, int n) {
    int i = blockIdx.x * 256 + threadIdx.x;
    if (i < n) out[i] = f2bf(in[i]);
}

// ---------------------------------------------------------------------------
// h[b,l,d] = x[b,l] * proj_w[d] + proj_b[d]   (f32), one float4 per thread
// ---------------------------------------------------------------------------
__global__ void __launch_bounds__(256)
proj_kernel(const float* __restrict__ x, const float* __restrict__ pw,
            const float* __restrict__ pb, float* __restrict__ h) {
    int idx = blockIdx.x * 256 + threadIdx.x;       // MROWS * (D/4) threads
    int d4  = idx & (DD / 4 - 1);
    int bl  = idx >> 7;                              // D/4 = 128
    float xv = x[bl];
    float4 w = *(const float4*)&pw[d4 * 4];
    float4 b = *(const float4*)&pb[d4 * 4];
    float4 r;
    r.x = xv * w.x + b.x; r.y = xv * w.y + b.y;
    r.z = xv * w.z + b.z; r.w = xv * w.w + b.w;
    *(float4*)&h[(size_t)bl * DD + d4 * 4] = r;
}

// ---------------------------------------------------------------------------
// LayerNorm over D=512: one wave32 per row (16 elements/lane), bf16 output
// ---------------------------------------------------------------------------
__global__ void __launch_bounds__(256)
ln_kernel(const float* __restrict__ h, const float* __restrict__ g,
          const float* __restrict__ b, __bf16* __restrict__ xn) {
    int row  = blockIdx.x * 8 + (threadIdx.x >> 5);
    int lane = threadIdx.x & 31;
    const float* hp = h + (size_t)row * DD;

    float4 vals[4];
    float sum = 0.f;
#pragma unroll
    for (int q = 0; q < 4; ++q) {
        vals[q] = *(const float4*)&hp[4 * lane + 128 * q];
        sum += vals[q].x + vals[q].y + vals[q].z + vals[q].w;
    }
#pragma unroll
    for (int off = 16; off >= 1; off >>= 1) sum += __shfl_xor(sum, off, 32);
    float mu = sum * (1.0f / DD);

    float vs = 0.f;
#pragma unroll
    for (int q = 0; q < 4; ++q) {
        float dx = vals[q].x - mu, dy = vals[q].y - mu;
        float dz = vals[q].z - mu, dw = vals[q].w - mu;
        vs += dx * dx + dy * dy + dz * dz + dw * dw;
    }
#pragma unroll
    for (int off = 16; off >= 1; off >>= 1) vs += __shfl_xor(vs, off, 32);
    float inv = rsqrtf(vs * (1.0f / DD) + EPSF);

    __bf16* xp = xn + (size_t)row * DD;
#pragma unroll
    for (int q = 0; q < 4; ++q) {
        int d = 4 * lane + 128 * q;
        float4 gg = *(const float4*)&g[d];
        float4 bb = *(const float4*)&b[d];
        v4bf o;
        o[0] = f2bf((vals[q].x - mu) * inv * gg.x + bb.x);
        o[1] = f2bf((vals[q].y - mu) * inv * gg.y + bb.y);
        o[2] = f2bf((vals[q].z - mu) * inv * gg.z + bb.z);
        o[3] = f2bf((vals[q].w - mu) * inv * gg.w + bb.w);
        *(v4bf*)&xp[d] = o;
    }
}

// ---------------------------------------------------------------------------
// Causal depthwise conv (K=4): c[b,l,d] = dw_b[d] + sum_j xn[b,l-3+j,d]*w[d,j]
// One thread = 8 consecutive channels of one (b,l).
// ---------------------------------------------------------------------------
__global__ void __launch_bounds__(256)
conv_kernel(const __bf16* __restrict__ xn, const float* __restrict__ w,
            const float* __restrict__ bias, __bf16* __restrict__ c) {
    int idx = blockIdx.x * 256 + threadIdx.x;       // MROWS * (D/8) threads
    int d8  = idx & (DD / 8 - 1);
    int bl  = idx >> 6;                              // D/8 = 64
    int l   = bl & (LL - 1);
    int d0  = d8 * 8;

    float acc[8];
#pragma unroll
    for (int t = 0; t < 8; ++t) acc[t] = bias[d0 + t];

#pragma unroll
    for (int j = 0; j < KK; ++j) {
        int r = l - (KK - 1) + j;
        if (r < 0) continue;
        v8bf xv = *(const v8bf*)&xn[(size_t)(bl - (KK - 1) + j) * DD + d0];
#pragma unroll
        for (int t = 0; t < 8; ++t)
            acc[t] += bf2f(xv[t]) * w[(d0 + t) * KK + j];
    }
    v8bf o;
#pragma unroll
    for (int t = 0; t < 8; ++t) o[t] = f2bf(acc[t]);
    *(v8bf*)&c[(size_t)bl * DD + d0] = o;
}

// ---------------------------------------------------------------------------
// WMMA GEMM: out(M x 512) = A(M x 512, bf16 row-major) @ W(512 x 512, bf16)
// 256 threads (8 wave32s), block tile 128x128, BK=32.
// Double-buffered LDS filled by GLOBAL_LOAD_ASYNC_TO_LDS_B128 (ASYNCcnt);
// DMA of tile k+1 overlaps WMMA compute on tile k; one barrier per k-step.
// Epilogue modes:
//   0: out_bf = bf16( silu(acc + bias[n]) )
//   1: out_bf = bf16( aux_bf[m,n] * sigmoid(acc + bias[n]) )
//   2: h_f32[m,n] += acc + bias[n]           (residual)
// ---------------------------------------------------------------------------
#define BM 128
#define BN 128
#define BK 32
#define LDA_S (BK + 8)     // 40 bf16 -> 80 B row stride (16B aligned)
#define LDB_S (BN + 8)     // 136 bf16 -> 272 B row stride (16B aligned)

template <int MODE>
__global__ void __launch_bounds__(256)
gemm_bf16_kernel(const __bf16* __restrict__ A, const __bf16* __restrict__ W,
                 const float* __restrict__ bias, const __bf16* __restrict__ aux,
                 float* __restrict__ hbuf, __bf16* __restrict__ outbf) {
    __shared__ __bf16 sA[2][BM * LDA_S];
    __shared__ __bf16 sB[2][BK * LDB_S];

    const int tid  = threadIdx.x;
    const int lane = tid & 31;
    const int wave = tid >> 5;
    const int wm   = wave & 1;            // 2 waves along M (64 rows each)
    const int wn   = wave >> 1;           // 4 waves along N (32 cols each)
    const int m_block = blockIdx.y * BM;
    const int n_block = blockIdx.x * BN;
    const int lml = lane & 15;
    const int lhi = lane >> 4;

    v8f acc[4][2];
#pragma unroll
    for (int i = 0; i < 4; ++i)
#pragma unroll
        for (int j = 0; j < 2; ++j)
#pragma unroll
            for (int v = 0; v < 8; ++v) acc[i][j][v] = 0.0f;

    // cooperative tile-load coordinates (16 bf16 = 32 B per thread per tile)
    const int a_row = tid >> 1;              // 0..127
    const int a_col = (tid & 1) * 16;        // 0 / 16
    const int b_row = tid >> 3;              // 0..31
    const int b_col = (tid & 7) * 16;        // 0..112

    const __bf16* gA = &A[(size_t)(m_block + a_row) * DD + a_col];
    const __bf16* gB = &W[(size_t)b_row * DD + n_block + b_col];

    // prologue: async-DMA tile 0 into buffer 0
    async_ld32B(&sA[0][a_row * LDA_S + a_col], gA);
    async_ld32B(&sB[0][b_row * LDB_S + b_col], gB);
    wait_async0();
    __syncthreads();

    int cur = 0;
    for (int k0 = 0; k0 < DD; k0 += BK) {
        const int nxt = k0 + BK;
        if (nxt < DD) {   // async-DMA next tile into shadow buffer (overlaps WMMA)
            async_ld32B(&sA[cur ^ 1][a_row * LDA_S + a_col], gA + nxt);
            async_ld32B(&sB[cur ^ 1][b_row * LDB_S + b_col], gB + (size_t)nxt * DD);
        }

        // B fragments: lane = col N, K = {8*lhi..+7, 16+8*lhi..+7}
        v16bf bfr[2];
#pragma unroll
        for (int j = 0; j < 2; ++j) {
            const __bf16* p = &sB[cur][8 * lhi * LDB_S + wn * 32 + j * 16 + lml];
#pragma unroll
            for (int e = 0; e < 8; ++e) {
                bfr[j][e] = p[e * LDB_S];
                bfr[j][e + 8] = p[(16 + e) * LDB_S];
            }
        }
#pragma unroll
        for (int i = 0; i < 4; ++i) {
            // A fragment: lane = row M, K = {8*lhi..+7, 16+8*lhi..+7}
            const __bf16* p = &sA[cur][(wm * 64 + i * 16 + lml) * LDA_S + 8 * lhi];
            v8bf lo = *(const v8bf*)p;
            v8bf hi = *(const v8bf*)(p + 16);
            v16bf afr = __builtin_shufflevector(lo, hi, 0, 1, 2, 3, 4, 5, 6, 7,
                                                8, 9, 10, 11, 12, 13, 14, 15);
#pragma unroll
            for (int j = 0; j < 2; ++j)
                acc[i][j] = __builtin_amdgcn_wmma_f32_16x16x32_bf16(
                    false, afr, false, bfr[j], (short)0, acc[i][j], false, false);
        }

        if (nxt < DD) {
            wait_async0();       // this wave's DMA into shadow buffer landed
            __syncthreads();     // every wave's DMA landed + reads of cur done
            cur ^= 1;
        }
    }

    // Epilogue. D layout: lanes 0-15 -> M = v, lanes 16-31 -> M = v+8; N = lane&15
#pragma unroll
    for (int i = 0; i < 4; ++i) {
#pragma unroll
        for (int j = 0; j < 2; ++j) {
            int n = n_block + wn * 32 + j * 16 + lml;
            float bv = bias[n];
#pragma unroll
            for (int v = 0; v < 8; ++v) {
                int m = m_block + wm * 64 + i * 16 + 8 * lhi + v;
                size_t off = (size_t)m * DD + n;
                float val = acc[i][j][v] + bv;
                if (MODE == 0) {
                    outbf[off] = f2bf(val / (1.0f + __expf(-val)));   // silu
                } else if (MODE == 1) {
                    float gg = 1.0f / (1.0f + __expf(-val));          // sigmoid
                    outbf[off] = f2bf(bf2f(aux[off]) * gg);
                } else {
                    hbuf[off] += val;                                  // residual
                }
            }
        }
    }
}

// ---------------------------------------------------------------------------
// Head: out[b] = dot(h[b, L-1, :], head_w) + head_b
// ---------------------------------------------------------------------------
__global__ void __launch_bounds__(256)
head_kernel(const float* __restrict__ h, const float* __restrict__ hw,
            const float* __restrict__ hb, float* __restrict__ out) {
    __shared__ float red[256];
    int b = blockIdx.x;
    const float* hp = h + ((size_t)b * LL + (LL - 1)) * DD;
    float s = 0.f;
    for (int d = threadIdx.x; d < DD; d += 256) s += hp[d] * hw[d];
    red[threadIdx.x] = s;
    __syncthreads();
#pragma unroll
    for (int st = 128; st > 0; st >>= 1) {
        if ((int)threadIdx.x < st) red[threadIdx.x] += red[threadIdx.x + st];
        __syncthreads();
    }
    if (threadIdx.x == 0) out[b] = red[0] + hb[0];
}

// ---------------------------------------------------------------------------
// Host launch
// ---------------------------------------------------------------------------
extern "C" void kernel_launch(void* const* d_in, const int* in_sizes, int n_in,
                              void* d_out, int out_size, void* d_ws, size_t ws_size,
                              hipStream_t stream) {
    const float* x      = (const float*)d_in[0];
    const float* proj_w = (const float*)d_in[1];
    const float* proj_b = (const float*)d_in[2];
    const float* norm_g = (const float*)d_in[3];
    const float* norm_b = (const float*)d_in[4];
    const float* dw_w   = (const float*)d_in[5];
    const float* dw_b   = (const float*)d_in[6];
    const float* pw1_w  = (const float*)d_in[7];
    const float* pw1_b  = (const float*)d_in[8];
    const float* gate_w = (const float*)d_in[9];
    const float* gate_b = (const float*)d_in[10];
    const float* out_w  = (const float*)d_in[11];
    const float* out_b  = (const float*)d_in[12];
    const float* head_w = (const float*)d_in[13];
    const float* head_b = (const float*)d_in[14];
    float* outp = (float*)d_out;

    const size_t HD = (size_t)MROWS * DD;         // 67,108,864 elems
    char* ws = (char*)d_ws;
    size_t off = 0;
    float*  hbuf = (float*)(ws + off);  off += HD * 4;
    __bf16* xn   = (__bf16*)(ws + off); off += HD * 2;
    __bf16* cbuf = (__bf16*)(ws + off); off += HD * 2;   // also reused as `u`
    __bf16* abuf = (__bf16*)(ws + off); off += HD * 2;
    __bf16* wpw1 = (__bf16*)(ws + off); off += (size_t)NL * DD * DD * 2;
    __bf16* wgat = (__bf16*)(ws + off); off += (size_t)NL * DD * DD * 2;
    __bf16* wout = (__bf16*)(ws + off); off += (size_t)NL * DD * DD * 2;

    // weights f32 -> bf16 (all layers at once)
    {
        int n = NL * DD * DD;
        int blocks = (n + 255) / 256;
        cvt_bf16_kernel<<<blocks, 256, 0, stream>>>(pw1_w,  wpw1, n);
        cvt_bf16_kernel<<<blocks, 256, 0, stream>>>(gate_w, wgat, n);
        cvt_bf16_kernel<<<blocks, 256, 0, stream>>>(out_w,  wout, n);
    }

    // h = x @ proj_w + proj_b
    proj_kernel<<<(MROWS * (DD / 4)) / 256, 256, 0, stream>>>(x, proj_w, proj_b, hbuf);

    dim3 gemm_grid(DD / BN, MROWS / BM);  // (4, 1024)
    for (int i = 0; i < NL; ++i) {
        ln_kernel<<<MROWS / 8, 256, 0, stream>>>(
            hbuf, norm_g + (size_t)i * DD, norm_b + (size_t)i * DD, xn);

        conv_kernel<<<(MROWS * (DD / 8)) / 256, 256, 0, stream>>>(
            xn, dw_w + (size_t)i * DD * KK, dw_b + (size_t)i * DD, cbuf);

        // a = silu(c @ pw1 + b1)
        gemm_bf16_kernel<0><<<gemm_grid, 256, 0, stream>>>(
            cbuf, wpw1 + (size_t)i * DD * DD, pw1_b + (size_t)i * DD,
            nullptr, nullptr, abuf);

        // u = a * sigmoid(xn @ gate + bg)   (written into cbuf, now free)
        gemm_bf16_kernel<1><<<gemm_grid, 256, 0, stream>>>(
            xn, wgat + (size_t)i * DD * DD, gate_b + (size_t)i * DD,
            abuf, nullptr, cbuf);

        // h += u @ out_w + out_b
        gemm_bf16_kernel<2><<<gemm_grid, 256, 0, stream>>>(
            cbuf, wout + (size_t)i * DD * DD, out_b + (size_t)i * DD,
            nullptr, hbuf, nullptr);
    }

    head_kernel<<<BB, 256, 0, stream>>>(hbuf, head_w, head_b, outp);
}
